// TimeDecoder_84043920048431
// MI455X (gfx1250) — matrix operand order repaired and checked by reading, not compiled, not run on hardware
//
#include <hip/hip_runtime.h>
#include <hip/hip_bf16.h>

// ---------------------------------------------------------------------------
// TimeDecoder on MI455X (gfx1250): WMMA f16 16x16x32 for all GEMMs, fused
// LDS-resident GRU recurrences, VALU conv-transpose stacks.
// ---------------------------------------------------------------------------

typedef __attribute__((ext_vector_type(16))) _Float16 v16h;
typedef __attribute__((ext_vector_type(8)))  float    v8f;

#define B_   128
#define LAT_ 64
#define H_   128
#define S_   256
#define F_   16
#define K_   5

// ---- CDNA5 WMMA fragment helpers (contiguous k order in A and B, wave32) ----------

// A (16xK, f16), row-major source with row stride lda (in elements).
// lane<16: M=lane, halves 0..15 -> k0+0..15
// lane>=16: M=lane-16, halves 0..15 -> k0+16..31
__device__ __forceinline__ v16h load_a_frag(const _Float16* A, long lda,
                                            int row0, int k0, int lane) {
  const int m    = lane & 15;
  const int koff = ((lane >> 4) & 1) * 16;
  const _Float16* p = A + (long)(row0 + m) * lda + k0 + koff;
  v16h r;
#pragma unroll
  for (int i = 0; i < 16; ++i) r[i] = p[i];
  return r;
}

// B (Kx16) sourced from weight stored (N,K) row-major (i.e. W, used as W^T):
// lane: n = n0 + (lane&15); halves i -> k = k0 + (lane>=16?16:0) + i  (contiguous!)
__device__ __forceinline__ v16h load_b_frag(const _Float16* W, long ldw,
                                            int n0, int k0, int lane) {
  const int n    = n0 + (lane & 15);
  const int koff = ((lane >> 4) & 1) * 16;
  const _Float16* p = W + (long)n * ldw + k0 + koff;
  v16h r;
#pragma unroll
  for (int i = 0; i < 16; ++i) r[i] = p[i];
  return r;
}

__device__ __forceinline__ v8f wmma_f16(v16h a, v16h b, v8f c) {
  // D = A*B + C ; f32 accumulate
  return __builtin_amdgcn_wmma_f32_16x16x32_f16(false, a, false, b,
                                                (short)0, c, false, false);
}

__device__ __forceinline__ float sigmoidf_(float x) {
  return 1.0f / (1.0f + __expf(-x));
}
__device__ __forceinline__ float leaky_(float v) {
  return v >= 0.0f ? v : 0.01f * v;
}

// ---- f32 -> f16 elementwise convert ----------------------------------------
__global__ void f32_to_f16_kernel(const float* __restrict__ src,
                                  _Float16* __restrict__ dst, int n) {
  int i = blockIdx.x * blockDim.x + threadIdx.x;
  if (i < n) dst[i] = (_Float16)src[i];
}

// ---- Generic WMMA GEMM: C(MxN) = A(MxK) * W(N,K)^T + bias ------------------
// One 16x16 tile per wave. M,N multiples of 16; K multiple of 32.
__global__ void gemm_bias_f16(const _Float16* __restrict__ A,
                              const _Float16* __restrict__ W,
                              const float* __restrict__ bias,
                              float* __restrict__ Cf, _Float16* __restrict__ Ch,
                              int M, int N, int K) {
  const int lane = threadIdx.x & 31;
  const int wave = blockIdx.x * (blockDim.x >> 5) + (threadIdx.x >> 5);
  const int tilesN = N >> 4;
  const int tM = wave / tilesN;
  const int tN = wave - tM * tilesN;
  if (tM * 16 >= M) return;  // wave-uniform

  v8f acc = {};
  for (int k0 = 0; k0 < K; k0 += 32) {
    v16h a = load_a_frag(A, K, tM * 16, k0, lane);
    v16h b = load_b_frag(W, K, tN * 16, k0, lane);
    acc = wmma_f16(a, b, acc);
  }
  const int n  = tN * 16 + (lane & 15);
  const float bv = bias ? bias[n] : 0.0f;
  const int mb = tM * 16 + ((lane >> 4) & 1) * 8;
#pragma unroll
  for (int v = 0; v < 8; ++v) {
    float val = acc[v] + bv;
    long idx = (long)(mb + v) * N + n;
    if (Cf) Cf[idx] = val;
    if (Ch) Ch[idx] = (_Float16)val;
  }
}

// ---- Fused bidirectional GRU (H=128, T=256, input 64) ----------------------
// grid = (8 batch-groups of 16, 2 directions); block = 256 threads (8 waves).
// Per step, wave w computes N-tiles w*48..w*48+47 of gi (x_s @ Wih^T) and
// gh (h @ Whh^T) with WMMA; VALU gate math; state in LDS.
__global__ void bigru_kernel(const _Float16* __restrict__ X1,   // (B*S, 64) f16
                             const float* __restrict__ Wih_f, const float* __restrict__ Whh_f,
                             const float* __restrict__ bih_f, const float* __restrict__ bhh_f,
                             const float* __restrict__ Wih_b, const float* __restrict__ Whh_b,
                             const float* __restrict__ bih_b, const float* __restrict__ bhh_b,
                             _Float16* __restrict__ X2) {      // (B*S, 256) f16
  extern __shared__ char smem[];
  _Float16* Wih_h = (_Float16*)smem;              // 384*64
  _Float16* Whh_h = Wih_h + 384 * 64;             // 384*128
  _Float16* h_h   = Whh_h + 384 * 128;            // 16*128 (state, f16)
  float* h_f  = (float*)(h_h + 16 * 128);         // 16*128 (state, f32)
  float* gi   = h_f + 16 * 128;                   // 16*384
  float* gh   = gi + 16 * 384;                    // 16*384
  float* bihs = gh + 16 * 384;                    // 384
  float* bhhs = bihs + 384;                       // 384

  const int tid  = threadIdx.x;
  const int lane = tid & 31;
  const int wave = tid >> 5;
  const int b0   = blockIdx.x * 16;
  const int fwd  = (blockIdx.y == 0);
  const int colOff = fwd ? 0 : H_;

  const float* Wih = fwd ? Wih_f : Wih_b;
  const float* Whh = fwd ? Whh_f : Whh_b;
  const float* bih = fwd ? bih_f : bih_b;
  const float* bhh = fwd ? bhh_f : bhh_b;

  for (int i = tid; i < 384 * 64;  i += 256) Wih_h[i] = (_Float16)Wih[i];
  for (int i = tid; i < 384 * 128; i += 256) Whh_h[i] = (_Float16)Whh[i];
  for (int i = tid; i < 384; i += 256) { bihs[i] = bih[i]; bhhs[i] = bhh[i]; }
  for (int i = tid; i < 16 * 128; i += 256) { h_f[i] = 0.0f; h_h[i] = (_Float16)0.0f; }
  __syncthreads();

  for (int t = 0; t < S_; ++t) {
    const int s = fwd ? t : (S_ - 1 - t);
    const _Float16* Xbase = X1 + (long)(b0 * S_ + s) * LAT_;   // row stride S_*LAT_

#pragma unroll
    for (int u = 0; u < 3; ++u) {
      const int n0 = wave * 48 + u * 16;
      v8f ai = {};
#pragma unroll
      for (int kc = 0; kc < 2; ++kc) {   // K = 64
        v16h a = load_a_frag(Xbase, (long)S_ * LAT_, 0, kc * 32, lane);
        v16h b = load_b_frag(Wih_h, LAT_, n0, kc * 32, lane);
        ai = wmma_f16(a, b, ai);
      }
      v8f ah = {};
#pragma unroll
      for (int kc = 0; kc < 4; ++kc) {   // K = 128
        v16h a = load_a_frag(h_h, H_, 0, kc * 32, lane);
        v16h b = load_b_frag(Whh_h, H_, n0, kc * 32, lane);
        ah = wmma_f16(a, b, ah);
      }
      const int n  = n0 + (lane & 15);
      const int mb = ((lane >> 4) & 1) * 8;
#pragma unroll
      for (int v = 0; v < 8; ++v) {
        gi[(mb + v) * 384 + n] = ai[v];
        gh[(mb + v) * 384 + n] = ah[v];
      }
    }
    __syncthreads();

    // gate math: 16 rows x 128 cols, 8 elems/thread
    {
      const int m  = tid >> 4;
      const int jb = (tid & 15) * 8;
#pragma unroll
      for (int q = 0; q < 8; ++q) {
        const int j = jb + q;
        float gr = gi[m * 384 + j] + bihs[j] + gh[m * 384 + j] + bhhs[j];
        float gz = gi[m * 384 + H_ + j] + bihs[H_ + j] + gh[m * 384 + H_ + j] + bhhs[H_ + j];
        float r  = sigmoidf_(gr);
        float z  = sigmoidf_(gz);
        float nn = tanhf(gi[m * 384 + 2 * H_ + j] + bihs[2 * H_ + j] +
                         r * (gh[m * 384 + 2 * H_ + j] + bhhs[2 * H_ + j]));
        float hn = (1.0f - z) * nn + z * h_f[m * H_ + j];
        h_f[m * H_ + j] = hn;
        h_h[m * H_ + j] = (_Float16)hn;
        X2[((long)(b0 + m) * S_ + s) * (2 * H_) + colOff + j] = (_Float16)hn;
      }
    }
    __syncthreads();
  }
}

// ---- Fused GRU-e (F=16, input 2H=256) --------------------------------------
// One workgroup (256 threads = 8 waves); wave w owns M-tile w (16 batch rows).
__global__ void gru_e_kernel(const _Float16* __restrict__ X2,  // (B*S, 256) f16
                             const float* __restrict__ Wih, const float* __restrict__ Whh,
                             const float* __restrict__ bih, const float* __restrict__ bhh,
                             float* __restrict__ cnn_in) {     // (B, F, S) f32
  extern __shared__ char smem[];
  _Float16* Wih_h = (_Float16*)smem;              // 48*256
  _Float16* Whh_h = Wih_h + 48 * 256;             // 48*32 (K padded 16->32)
  _Float16* h_h   = Whh_h + 48 * 32;              // 128*32 (K padded)
  float* h_f  = (float*)(h_h + 128 * 32);         // 128*16
  float* gi   = h_f + 128 * 16;                   // 128*48
  float* gh   = gi + 128 * 48;                    // 128*48
  float* bihs = gh + 128 * 48;                    // 48
  float* bhhs = bihs + 48;                        // 48

  const int tid = threadIdx.x, lane = tid & 31, wave = tid >> 5;

  for (int i = tid; i < 48 * 256; i += 256) Wih_h[i] = (_Float16)Wih[i];
  for (int i = tid; i < 48 * 32; i += 256) {
    int n = i >> 5, k = i & 31;
    Whh_h[i] = (k < F_) ? (_Float16)Whh[n * F_ + k] : (_Float16)0.0f;
  }
  for (int i = tid; i < 128 * 32; i += 256) h_h[i] = (_Float16)0.0f;
  for (int i = tid; i < 128 * 16; i += 256) h_f[i] = 0.0f;
  if (tid < 48) { bihs[tid] = bih[tid]; bhhs[tid] = bhh[tid]; }
  __syncthreads();

  for (int s = 0; s < S_; ++s) {
    const int m0 = wave * 16;
    const _Float16* Xbase = X2 + (long)s * (2 * H_);  // row stride S_*2H
#pragma unroll
    for (int u = 0; u < 3; ++u) {
      const int n0 = u * 16;
      v8f ai = {};
#pragma unroll
      for (int kc = 0; kc < 8; ++kc) {  // K = 256
        v16h a = load_a_frag(Xbase, (long)S_ * (2 * H_), m0, kc * 32, lane);
        v16h b = load_b_frag(Wih_h, 2 * H_, n0, kc * 32, lane);
        ai = wmma_f16(a, b, ai);
      }
      v8f ah = {};
      {
        v16h a = load_a_frag(h_h, 32, m0, 0, lane);
        v16h b = load_b_frag(Whh_h, 32, n0, 0, lane);
        ah = wmma_f16(a, b, ah);
      }
      const int n  = n0 + (lane & 15);
      const int mb = m0 + ((lane >> 4) & 1) * 8;
#pragma unroll
      for (int v = 0; v < 8; ++v) {
        gi[(mb + v) * 48 + n] = ai[v];
        gh[(mb + v) * 48 + n] = ah[v];
      }
    }
    __syncthreads();

    {
      const int m  = tid >> 1;           // batch row 0..127
      const int jb = (tid & 1) * 8;
#pragma unroll
      for (int q = 0; q < 8; ++q) {
        const int j = jb + q;
        float gr = gi[m * 48 + j] + bihs[j] + gh[m * 48 + j] + bhhs[j];
        float gz = gi[m * 48 + F_ + j] + bihs[F_ + j] + gh[m * 48 + F_ + j] + bhhs[F_ + j];
        float r  = sigmoidf_(gr);
        float z  = sigmoidf_(gz);
        float nn = tanhf(gi[m * 48 + 2 * F_ + j] + bihs[2 * F_ + j] +
                         r * (gh[m * 48 + 2 * F_ + j] + bhhs[2 * F_ + j]));
        float hn = (1.0f - z) * nn + z * h_f[m * F_ + j];
        h_f[m * F_ + j] = hn;
        h_h[m * 32 + j] = (_Float16)hn;
        cnn_in[(long)m * (F_ * S_) + j * S_ + s] = hn;  // (B,F,S)
      }
    }
    __syncthreads();
  }
}

// ---- Per-(batch,feature) conv-transpose stack 1->32->64->1, LeakyReLU ------
// conv_t(stride1,pad2) == y[co][p] = b + sum_{ci,j=-2..2} x[ci][p+j]*w[ci][co][2-j]
__global__ void cnn_stack_kernel(const float* __restrict__ in,   // (B,F,S)
                                 const float* __restrict__ w3, const float* __restrict__ b3,
                                 const float* __restrict__ w2, const float* __restrict__ b2,
                                 const float* __restrict__ w1, const float* __restrict__ b1,
                                 float* __restrict__ out, int transpose_out) {
  extern __shared__ char smem[];
  float* xin = (float*)smem;          // 256
  float* h1  = xin + 256;             // 32*256
  float* h2  = h1 + 32 * 256;         // 64*256
  float* w3s = h2 + 64 * 256;         // 32*5
  float* w2s = w3s + 160;             // 32*64*5
  float* w1s = w2s + 10240;           // 64*5
  float* b3s = w1s + 320;             // 32
  float* b2s = b3s + 32;              // 64

  const int tid = threadIdx.x;
  const int b = blockIdx.x >> 4;
  const int f = blockIdx.x & 15;

  if (tid < 160) w3s[tid] = w3[f * 160 + tid];
  for (int i = tid; i < 10240; i += 256) w2s[i] = w2[f * 10240 + i];
  for (int i = tid; i < 320; i += 256) w1s[i] = w1[f * 320 + i];
  if (tid < 32) b3s[tid] = b3[f * 32 + tid];
  if (tid < 64) b2s[tid] = b2[f * 64 + tid];
  const float b1v = b1[f];
  xin[tid] = in[(long)b * (F_ * S_) + f * S_ + tid];
  __syncthreads();

  const int p = tid;
  // stage 1: 1 -> 32
  for (int co = 0; co < 32; ++co) {
    float acc = b3s[co];
#pragma unroll
    for (int k = 0; k < K_; ++k) {
      int q = p + k - 2;
      float xv = (q >= 0 && q < S_) ? xin[q] : 0.0f;
      acc += xv * w3s[co * K_ + (K_ - 1 - k)];
    }
    h1[co * S_ + p] = leaky_(acc);
  }
  __syncthreads();
  // stage 2: 32 -> 64
  for (int co = 0; co < 64; ++co) {
    float acc = b2s[co];
    for (int ci = 0; ci < 32; ++ci) {
      const float* wrow = &w2s[ci * 320 + co * K_];
#pragma unroll
      for (int k = 0; k < K_; ++k) {
        int q = p + k - 2;
        float xv = (q >= 0 && q < S_) ? h1[ci * S_ + q] : 0.0f;
        acc += xv * wrow[K_ - 1 - k];
      }
    }
    h2[co * S_ + p] = leaky_(acc);
  }
  __syncthreads();
  // stage 3: 64 -> 1
  {
    float acc = b1v;
    for (int ci = 0; ci < 64; ++ci) {
#pragma unroll
      for (int k = 0; k < K_; ++k) {
        int q = p + k - 2;
        float xv = (q >= 0 && q < S_) ? h2[ci * S_ + q] : 0.0f;
        acc += xv * w1s[ci * K_ + (K_ - 1 - k)];
      }
    }
    float o = leaky_(acc);
    if (transpose_out) out[(long)b * (S_ * F_) + p * F_ + f] = o;   // (B,S,F)
    else               out[(long)b * (F_ * S_) + f * S_ + p] = o;   // (B,F,S)
  }
}

// ---------------------------------------------------------------------------
extern "C" void kernel_launch(void* const* d_in, const int* in_sizes, int n_in,
                              void* d_out, int out_size, void* d_ws, size_t ws_size,
                              hipStream_t stream) {
  (void)in_sizes; (void)n_in; (void)out_size; (void)ws_size;
  const float* x     = (const float*)d_in[0];
  const float* Ws    = (const float*)d_in[1];
  const float* bs    = (const float*)d_in[2];
  const float* Wih_f = (const float*)d_in[3];
  const float* Whh_f = (const float*)d_in[4];
  const float* bih_f = (const float*)d_in[5];
  const float* bhh_f = (const float*)d_in[6];
  const float* Wih_b = (const float*)d_in[7];
  const float* Whh_b = (const float*)d_in[8];
  const float* bih_b = (const float*)d_in[9];
  const float* bhh_b = (const float*)d_in[10];
  const float* Wih_e = (const float*)d_in[11];
  const float* Whh_e = (const float*)d_in[12];
  const float* bih_e = (const float*)d_in[13];
  const float* bhh_e = (const float*)d_in[14];
  const float* w3    = (const float*)d_in[15];
  const float* b3    = (const float*)d_in[16];
  const float* w2    = (const float*)d_in[17];
  const float* b2    = (const float*)d_in[18];
  const float* w1    = (const float*)d_in[19];
  const float* b1    = (const float*)d_in[20];
  float* outp = (float*)d_out;

  char* ws = (char*)d_ws;
  size_t off = 0;
  auto alloc = [&](size_t bytes) -> void* {
    void* p = ws + off;
    off += (bytes + 255) & ~(size_t)255;
    return p;
  };
  _Float16* x_h   = (_Float16*)alloc((size_t)B_ * LAT_ * 2);
  _Float16* Ws_h  = (_Float16*)alloc((size_t)LAT_ * S_ * LAT_ * 2);      // 16384x64
  _Float16* X1_h  = (_Float16*)alloc((size_t)B_ * S_ * LAT_ * 2);        // (B*S,64)
  _Float16* X2_h  = (_Float16*)alloc((size_t)B_ * S_ * 2 * H_ * 2);      // (B*S,256)
  float* cnn_in   = (float*)alloc((size_t)B_ * F_ * S_ * 4);
  float* cnn_mid  = (float*)alloc((size_t)B_ * F_ * S_ * 4);

  // 1) convert x and Ws to f16
  f32_to_f16_kernel<<<(B_ * LAT_ + 255) / 256, 256, 0, stream>>>(x, x_h, B_ * LAT_);
  f32_to_f16_kernel<<<(LAT_ * S_ * LAT_ + 255) / 256, 256, 0, stream>>>(Ws, Ws_h, LAT_ * S_ * LAT_);

  // 2) h0 = x @ Ws^T + bs   -> X1 (f16, (B*S, LAT) via row-major reinterpret)
  //    M=128, N=16384, K=64 -> 8*1024 = 8192 tiles, 8 waves/block -> 1024 blocks
  gemm_bias_f16<<<1024, 256, 0, stream>>>(x_h, Ws_h, bs, nullptr, X1_h,
                                          B_, LAT_ * S_, LAT_);

  // 3) bidirectional GRU (fwd: blockIdx.y==0 writes cols 0..127; bwd: 128..255)
  size_t shm_bigru = (size_t)(384 * 64 + 384 * 128 + 16 * 128) * sizeof(_Float16)
                   + (size_t)(16 * 128 + 2 * 16 * 384 + 2 * 384) * sizeof(float);
  bigru_kernel<<<dim3(8, 2), 256, shm_bigru, stream>>>(
      X1_h, Wih_f, Whh_f, bih_f, bhh_f, Wih_b, Whh_b, bih_b, bhh_b, X2_h);

  // 4) GRU-e -> cnn_in (B,F,S)
  size_t shm_e = (size_t)(48 * 256 + 48 * 32 + 128 * 32) * sizeof(_Float16)
               + (size_t)(128 * 16 + 2 * 128 * 48 + 2 * 48) * sizeof(float);
  gru_e_kernel<<<1, 256, shm_e, stream>>>(X2_h, Wih_e, Whh_e, bih_e, bhh_e, cnn_in);

  // 5) conv-transpose stack, twice; second pass writes (B,S,F) to d_out
  size_t shm_cnn = (size_t)(256 + 32 * 256 + 64 * 256 + 160 + 10240 + 320 + 32 + 64)
                 * sizeof(float);
  cnn_stack_kernel<<<B_ * F_, 256, shm_cnn, stream>>>(cnn_in, w3, b3, w2, b2, w1, b1,
                                                      cnn_mid, 0);
  cnn_stack_kernel<<<B_ * F_, 256, shm_cnn, stream>>>(cnn_mid, w3, b3, w2, b2, w1, b1,
                                                      outp, 1);
}
